// GraphNN_75539884802616
// MI455X (gfx1250) — compile-verified, matrix operand order checked by reading, not run on hardware
//
#include <hip/hip_runtime.h>

// ---------------- problem constants (match reference) ----------------
#define BGRAPH 64
#define N_PER  1024
#define F_IN   64
#define HID    128
#define NNODES (BGRAPH * N_PER)      // 65536
#define NEDGES (NNODES * 16)         // 1048576
#define K1     820                   // ceil(0.8*1024)
#define K2     656                   // ceil(0.8*820)
#define K3     525                   // ceil(0.8*656)

typedef __attribute__((ext_vector_type(16))) _Float16 v16h;
typedef __attribute__((ext_vector_type(8)))  float    v8f;

// ---------------- utility ----------------
__global__ void zero_f32(float* p, int n) {
  int i = blockIdx.x * blockDim.x + threadIdx.x;
  if (i < n) p[i] = 0.0f;
}

__global__ void init_edges(const int* __restrict__ eidx, int* __restrict__ src,
                           int* __restrict__ dst, float* __restrict__ em, int E) {
  int e = blockIdx.x * blockDim.x + threadIdx.x;
  if (e >= E) return;
  src[e] = eidx[e];          // edge_index[0, e]
  dst[e] = eidx[E + e];      // edge_index[1, e]
  em[e]  = 1.0f;
}

// ---------------- edge aggregation: agg[dst] += x[src] * emask ----------------
// One thread per (edge, 4-feature chunk): coalesced gather + coalesced atomics.
template <int K>
__global__ void edge_agg(const float* __restrict__ x, const int* __restrict__ src,
                         const int* __restrict__ dst, const float* __restrict__ em,
                         float* __restrict__ agg, int total /* = E*K/4 */) {
  constexpr int CHUNKS = K / 4;          // threads per edge
  int i = blockIdx.x * blockDim.x + threadIdx.x;
  if (i >= total) return;
  int e = i / CHUNKS;                    // power-of-two -> shift
  int c = i - e * CHUNKS;
  float m = em[e];
  if (m == 0.0f) return;
  const float4 v = *(const float4*)(x + (size_t)src[e] * K + c * 4);
  float* ad = agg + (size_t)dst[e] * K + c * 4;
  atomicAdd(ad + 0, v.x * m);
  atomicAdd(ad + 1, v.y * m);
  atomicAdd(ad + 2, v.z * m);
  atomicAdd(ad + 3, v.w * m);
}

// ---------------- conv GEMM with WMMA ----------------
// out[M,128] = relu(X[M,K] @ Wr[K,128] + AGG[M,K] @ Wn[K,128] + bias)
// block = 256 threads = 8 waves; block covers 16 rows, each wave one 16x16 tile.
template <int K>
__global__ void conv_wmma(const float* __restrict__ X, const float* __restrict__ AGG,
                          const float* __restrict__ Wr, const float* __restrict__ Wn,
                          const float* __restrict__ bias, float* __restrict__ out) {
  __shared__ _Float16 lA[2][16 * K];     // staged f16 A tiles (X and AGG)
  const int rowbase = blockIdx.x * 16;
  const int wave = threadIdx.x >> 5;
  const int lane = threadIdx.x & 31;
  const int colbase = wave * 16;
  const int lm = lane & 15;
  const int hi = lane >> 4;

  // cooperative, coalesced stage of the 16xK A tiles into LDS (float4 -> 4x f16)
  constexpr int ITEMS = 16 * K / 4;      // float4 items
#pragma unroll
  for (int it = threadIdx.x; it < ITEMS; it += 256) {
    int i4 = it * 4;
    int r = i4 / K, kk = i4 & (K - 1);
    size_t gi = (size_t)(rowbase + r) * K + kk;
    float4 vx = *(const float4*)(X + gi);
    float4 va = *(const float4*)(AGG + gi);
    lA[0][i4 + 0] = (_Float16)vx.x; lA[0][i4 + 1] = (_Float16)vx.y;
    lA[0][i4 + 2] = (_Float16)vx.z; lA[0][i4 + 3] = (_Float16)vx.w;
    lA[1][i4 + 0] = (_Float16)va.x; lA[1][i4 + 1] = (_Float16)va.y;
    lA[1][i4 + 2] = (_Float16)va.z; lA[1][i4 + 3] = (_Float16)va.w;
  }
  __syncthreads();

  const int n = colbase + lm;
  v8f c = {};
#pragma unroll
  for (int t = 0; t < 2; ++t) {
    const float* W = t ? Wn : Wr;
#pragma unroll
    for (int kb = 0; kb < K; kb += 32) {
      v16h a, b;
#pragma unroll
      for (int v = 0; v < 8; ++v) {
        // ISA 16-bit A layout: lanes 0-15 -> M rows; VGPR v holds K pair:
        int kk = ((v >> 2) << 4) + (hi << 3) + ((v & 3) << 1);  // 0..30
        int k0 = kb + kk;
        a[2 * v]     = lA[t][lm * K + k0];
        a[2 * v + 1] = lA[t][lm * K + k0 + 1];
        b[2 * v]     = (_Float16)W[(size_t)k0 * HID + n];
        b[2 * v + 1] = (_Float16)W[(size_t)(k0 + 1) * HID + n];
      }
      c = __builtin_amdgcn_wmma_f32_16x16x32_f16(
          /*neg_a=*/false, a, /*neg_b=*/false, b,
          /*c_mod=*/(short)0, c, /*reuse_a=*/false, /*reuse_b=*/false);
    }
  }

  const float bn = bias[n];
#pragma unroll
  for (int r = 0; r < 8; ++r) {
    // ISA 32-bit C/D layout: lanes 0-15: M=r; lanes 16-31: M=r+8
    int mm = rowbase + r + (hi << 3);
    float v = c[r] + bn;
    out[(size_t)mm * HID + n] = v > 0.0f ? v : 0.0f;
  }
}

// ---------------- pooling score: s[i] = (x[i].p) / ||p|| ----------------
__global__ void scores_k(const float* __restrict__ x, const float* __restrict__ p,
                         float* __restrict__ s, int M) {
  int i = blockIdx.x * blockDim.x + threadIdx.x;
  if (i >= M) return;
  float dot = 0.0f, pp = 0.0f;
  const float* xi = x + (size_t)i * HID;
#pragma unroll 8
  for (int f = 0; f < HID; ++f) { dot += xi[f] * p[f]; pp += p[f] * p[f]; }
  s[i] = dot * rsqrtf(pp);
}

// ---------------- per-graph top-k via in-LDS bitonic sort (descending) --------
__global__ void topk_pool(const float* __restrict__ s, float* __restrict__ scale,
                          int* __restrict__ perm, int* __restrict__ inv,
                          int k_prev, int k_new) {
  __shared__ float sv[1024];
  __shared__ int   si[1024];
  const int g = blockIdx.x;
  for (int i = threadIdx.x; i < 1024; i += blockDim.x) {
    sv[i] = (i < k_prev) ? s[g * k_prev + i] : -__builtin_inff();
    si[i] = i;
  }
  __syncthreads();
  for (int sz = 2; sz <= 1024; sz <<= 1) {
    for (int stride = sz >> 1; stride > 0; stride >>= 1) {
      for (int i = threadIdx.x; i < 512; i += blockDim.x) {
        int pos = 2 * i - (i & (stride - 1));
        int j = pos + stride;
        bool desc = ((pos & sz) == 0);
        float a = sv[pos], b = sv[j];
        bool sw = desc ? (a < b) : (a > b);
        if (sw) {
          sv[pos] = b; sv[j] = a;
          int t = si[pos]; si[pos] = si[j]; si[j] = t;
        }
      }
      __syncthreads();
    }
  }
  for (int i = threadIdx.x; i < k_prev; i += blockDim.x) inv[g * k_prev + i] = -1;
  __syncthreads();
  for (int j = threadIdx.x; j < k_new; j += blockDim.x) {
    int oldGlob = g * k_prev + si[j];
    int nid = g * k_new + j;
    perm[nid] = oldGlob;
    scale[nid] = tanhf(sv[j]);
    inv[oldGlob] = nid;
  }
}

// ---------------- x_new = x[perm] * tanh(score[perm]) ----------------
__global__ void gather_scale(const float* __restrict__ xin, const int* __restrict__ perm,
                             const float* __restrict__ scale, float* __restrict__ xout,
                             int total) {
  int i = blockIdx.x * blockDim.x + threadIdx.x;
  if (i >= total) return;
  int nid = i >> 7;           // /128
  int f = i & 127;
  xout[i] = xin[(size_t)perm[nid] * HID + f] * scale[nid];
}

// ---------------- relabel + filter edges ----------------
__global__ void relabel(int* __restrict__ src, int* __restrict__ dst,
                        float* __restrict__ em, const int* __restrict__ inv, int E) {
  int e = blockIdx.x * blockDim.x + threadIdx.x;
  if (e >= E) return;
  int ns = inv[src[e]], nd = inv[dst[e]];
  em[e] = em[e] * ((ns >= 0 && nd >= 0) ? 1.0f : 0.0f);
  src[e] = ns >= 0 ? ns : 0;
  dst[e] = nd >= 0 ? nd : 0;
}

// ---------------- readout: h[g] += [segmax | segmean] ----------------
__global__ void readout(const float* __restrict__ x, float* __restrict__ h, int k) {
  int g = blockIdx.x;
  int f = threadIdx.x;   // blockDim = 128
  float mx = -__builtin_inff(), sm = 0.0f;
  const float* xg = x + (size_t)g * k * HID;
  for (int j = 0; j < k; ++j) {
    float v = xg[(size_t)j * HID + f];
    mx = fmaxf(mx, v);
    sm += v;
  }
  h[g * 256 + f]       += mx;
  h[g * 256 + 128 + f] += sm / (float)k;
}

// ---------------- MLP head + log_softmax ----------------
__global__ void head_k(const float* __restrict__ h, const float* __restrict__ l1w,
                       const float* __restrict__ l1b, const float* __restrict__ l2w,
                       const float* __restrict__ l2b, float* __restrict__ out) {
  int g = blockIdx.x * blockDim.x + threadIdx.x;
  if (g >= BGRAPH) return;
  float hid[HID];
  const float* hg = h + g * 256;
  for (int j = 0; j < HID; ++j) {
    float acc = l1b[j];
    for (int i = 0; i < 256; ++i) acc += hg[i] * l1w[i * HID + j];
    hid[j] = acc > 0.0f ? acc : 0.0f;
  }
  float o0 = l2b[0], o1 = l2b[1];
  for (int i = 0; i < HID; ++i) {
    o0 += hid[i] * l2w[i * 2 + 0];
    o1 += hid[i] * l2w[i * 2 + 1];
  }
  o0 = o0 > 0.0f ? o0 : 0.0f;
  o1 = o1 > 0.0f ? o1 : 0.0f;
  float mx = fmaxf(o0, o1);
  float lse = mx + logf(expf(o0 - mx) + expf(o1 - mx));
  out[g * 2 + 0] = o0 - lse;
  out[g * 2 + 1] = o1 - lse;
}

// ---------------- host orchestration ----------------
extern "C" void kernel_launch(void* const* d_in, const int* in_sizes, int n_in,
                              void* d_out, int out_size, void* d_ws, size_t ws_size,
                              hipStream_t stream) {
  (void)in_sizes; (void)n_in; (void)out_size; (void)ws_size;
  const float* x_in = (const float*)d_in[0];
  const int*   eidx = (const int*)d_in[1];
  const float* Wr1 = (const float*)d_in[3],  *Wn1 = (const float*)d_in[4],  *b1 = (const float*)d_in[5];
  const float* Wr2 = (const float*)d_in[6],  *Wn2 = (const float*)d_in[7],  *b2 = (const float*)d_in[8];
  const float* Wr3 = (const float*)d_in[9],  *Wn3 = (const float*)d_in[10], *b3 = (const float*)d_in[11];
  const float* p1  = (const float*)d_in[12], *p2  = (const float*)d_in[13], *p3 = (const float*)d_in[14];
  const float* l1w = (const float*)d_in[15], *l1b = (const float*)d_in[16];
  const float* l2w = (const float*)d_in[17], *l2b = (const float*)d_in[18];
  float* out = (float*)d_out;

  // workspace carve-out (~114 MB)
  char* ws = (char*)d_ws;
  size_t off = 0;
  auto carve = [&](size_t bytes) -> void* {
    void* p = ws + off;
    off += (bytes + 255) & ~(size_t)255;
    return p;
  };
  float* bufA  = (float*)carve((size_t)NNODES * HID * 4);  // pooled x (layer input)
  float* bufB  = (float*)carve((size_t)NNODES * HID * 4);  // conv output
  float* agg   = (float*)carve((size_t)NNODES * HID * 4);  // edge aggregation
  float* sc    = (float*)carve((size_t)NNODES * 4);
  int*   perm  = (int*)  carve((size_t)NNODES * 4);
  float* scl   = (float*)carve((size_t)NNODES * 4);
  int*   inv   = (int*)  carve((size_t)NNODES * 4);
  int*   srcw  = (int*)  carve((size_t)NEDGES * 4);
  int*   dstw  = (int*)  carve((size_t)NEDGES * 4);
  float* emw   = (float*)carve((size_t)NEDGES * 4);
  float* hbuf  = (float*)carve((size_t)BGRAPH * 256 * 4);

  const int T = 256;
  const int egrid = (NEDGES + T - 1) / T;

  init_edges<<<egrid, T, 0, stream>>>(eidx, srcw, dstw, emw, NEDGES);
  zero_f32<<<(BGRAPH * 256 + T - 1) / T, T, 0, stream>>>(hbuf, BGRAPH * 256);

  struct Layer { int M, K, k_prev, k_new; const float *Wr, *Wn, *bb, *pv; };
  Layer L[3] = {
    { NNODES,      F_IN, N_PER, K1, Wr1, Wn1, b1, p1 },
    { BGRAPH * K1, HID,  K1,    K2, Wr2, Wn2, b2, p2 },
    { BGRAPH * K2, HID,  K2,    K3, Wr3, Wn3, b3, p3 },
  };

  for (int li = 0; li < 3; ++li) {
    const Layer& l = L[li];
    const float* xprev = (li == 0) ? x_in : bufA;

    // 1) aggregate messages along (masked, relabeled) edges
    int aggn = l.M * l.K;
    zero_f32<<<(aggn + T - 1) / T, T, 0, stream>>>(agg, aggn);
    if (l.K == F_IN) {
      int tot = NEDGES * (F_IN / 4);
      edge_agg<F_IN><<<(tot + T - 1) / T, T, 0, stream>>>(xprev, srcw, dstw, emw, agg, tot);
    } else {
      int tot = NEDGES * (HID / 4);
      edge_agg<HID><<<(tot + T - 1) / T, T, 0, stream>>>(xprev, srcw, dstw, emw, agg, tot);
    }

    // 2) conv GEMM (WMMA): bufB = relu(xprev@Wr + agg@Wn + b)
    if (l.K == F_IN)
      conv_wmma<F_IN><<<l.M / 16, 256, 0, stream>>>(xprev, agg, l.Wr, l.Wn, l.bb, bufB);
    else
      conv_wmma<HID><<<l.M / 16, 256, 0, stream>>>(xprev, agg, l.Wr, l.Wn, l.bb, bufB);

    // 3) top-k pooling
    scores_k<<<(l.M + T - 1) / T, T, 0, stream>>>(bufB, l.pv, sc, l.M);
    topk_pool<<<BGRAPH, 512, 0, stream>>>(sc, scl, perm, inv, l.k_prev, l.k_new);
    int tot = BGRAPH * l.k_new * HID;
    gather_scale<<<(tot + T - 1) / T, T, 0, stream>>>(bufB, perm, scl, bufA, tot);
    relabel<<<egrid, T, 0, stream>>>(srcw, dstw, emw, inv, NEDGES);

    // 4) readout accumulate
    readout<<<BGRAPH, HID, 0, stream>>>(bufA, hbuf, l.k_new);
  }

  head_k<<<1, BGRAPH, 0, stream>>>(hbuf, l1w, l1b, l2w, l2b, out);
}
